// ReadState_50457275794063
// MI455X (gfx1250) — matrix operand order repaired and checked by reading, not compiled, not run on hardware
//
#include <hip/hip_runtime.h>

typedef unsigned int  u32;
typedef unsigned short u16;
typedef __attribute__((ext_vector_type(16))) __bf16 v16bf;
typedef __attribute__((ext_vector_type(8)))  float  v8f;
typedef __attribute__((ext_vector_type(4)))  float  v4f;
typedef __attribute__((ext_vector_type(4)))  u32    v4u;

#define BATCH 128
#define NBANK 8
#define BM 128
#define BN 64
#define BK 32
#define LDA 40   // padded LDS stride (u16 elems): 32 + 8 pad -> conflict-free frag gathers
#define LDB 40

__device__ __forceinline__ u16 f2bf(float x) {
  u32 u = __builtin_bit_cast(u32, x);
  u32 r = u + 0x7FFFu + ((u >> 16) & 1u);   // round-to-nearest-even
  return (u16)(r >> 16);
}

// C(128 x N) += A_aug(128 x K) * Bmat(K x N), K = 8*A_COLS,
// A_aug[b, kb*A_COLS + d] = sel[b,kb] * Asrc[b,d].
// Split-K over gridDim.y with f32 HW atomics; bias added by the k==0 chunk.
// A_COLS is compile-time so bank tracking is divide-free.
template <int A_COLS>
__global__ __launch_bounds__(256) void banked_gemm_kernel(
    const float* __restrict__ Asrc, const float* __restrict__ sel,
    const float* __restrict__ Bmat, const float* __restrict__ bias,
    float* __restrict__ Cout, int N, int k_chunk)
{
  __shared__ __align__(16) u16 sA[BM * LDA];
  __shared__ __align__(16) u16 sB[BN * LDB];

  const int tid  = threadIdx.x;
  const int lane = tid & 31;
  const int wave = tid >> 5;
  const int wm   = wave & 3;          // wave row-tile  (4 x 32 rows)
  const int wn   = wave >> 2;         // wave col-tile  (2 x 32 cols)
  const int n0   = blockIdx.x * BN;
  const int kbeg = blockIdx.y * k_chunk;
  const int kend = kbeg + k_chunk;

  v8f acc[2][2];
  #pragma unroll
  for (int i = 0; i < 2; ++i)
    #pragma unroll
    for (int j = 0; j < 2; ++j) acc[i][j] = (v8f)0.0f;

  const int arow  = lane & 15;
  const int kpart = lane >> 4;        // 0/1 half-wave
  // A tile global-load mapping: 2 threads per row, 16 f32 each
  const int a_r  = tid >> 1;
  const int a_c0 = (tid & 1) * 16;
  // B tile global-load mapping: 8 threads per K-row, 8 f32 each
  const int b_k  = tid >> 3;
  const int b_n  = (tid & 7) * 8;

  // divide-free bank tracking (A_COLS is a power-of-two constant)
  int bank  = kbeg / A_COLS;
  int dbase = kbeg & (A_COLS - 1);
  float s = sel[a_r * NBANK + bank];
  const float* aptr = Asrc + (size_t)a_r * A_COLS + a_c0;

  for (int k0 = kbeg; k0 < kend; k0 += BK) {
    // ---- stage A tile (scale by sel, cvt bf16) ----
    {
      const v4f* src = reinterpret_cast<const v4f*>(aptr + dbase);
      #pragma unroll
      for (int i = 0; i < 4; ++i) {
        v4f f = src[i];
        uint2 pk;
        pk.x = (u32)f2bf(f.x * s) | ((u32)f2bf(f.y * s) << 16);
        pk.y = (u32)f2bf(f.z * s) | ((u32)f2bf(f.w * s) << 16);
        *reinterpret_cast<uint2*>(&sA[a_r * LDA + a_c0 + i * 4]) = pk;
      }
    }
    // ---- stage B tile transposed: sB[n][k] (cvt bf16, non-temporal stream) ----
    {
      const v4f* src = reinterpret_cast<const v4f*>(
          &Bmat[(size_t)(k0 + b_k) * N + n0 + b_n]);
      v4f f0 = __builtin_nontemporal_load(src);
      v4f f1 = __builtin_nontemporal_load(src + 1);
      sB[(b_n + 0) * LDB + b_k] = f2bf(f0.x);
      sB[(b_n + 1) * LDB + b_k] = f2bf(f0.y);
      sB[(b_n + 2) * LDB + b_k] = f2bf(f0.z);
      sB[(b_n + 3) * LDB + b_k] = f2bf(f0.w);
      sB[(b_n + 4) * LDB + b_k] = f2bf(f1.x);
      sB[(b_n + 5) * LDB + b_k] = f2bf(f1.y);
      sB[(b_n + 6) * LDB + b_k] = f2bf(f1.z);
      sB[(b_n + 7) * LDB + b_k] = f2bf(f1.w);
      if (k0 + BK < kend)   // pull next B tile toward the caches
        __builtin_prefetch(&Bmat[(size_t)(k0 + BK + b_k) * N + n0 + b_n], 0, 3);
    }
    __syncthreads();

    // ---- gather fragments (ISA bf16 A/B lane layouts) and WMMA ----
    v16bf aF[2], bF[2];
    #pragma unroll
    for (int mi = 0; mi < 2; ++mi) {
      const u16* base = &sA[(wm * 32 + mi * 16 + arow) * LDA];
      v4u lo = *reinterpret_cast<const v4u*>(base + 8 * kpart);        // K = 8*kpart..+7
      v4u hi = *reinterpret_cast<const v4u*>(base + 16 + 8 * kpart);   // K = 16+8*kpart..+7
      aF[mi] = __builtin_bit_cast(v16bf,
               __builtin_shufflevector(lo, hi, 0, 1, 2, 3, 4, 5, 6, 7));
    }
    #pragma unroll
    for (int ni = 0; ni < 2; ++ni) {
      const u16* base = &sB[(wn * 32 + ni * 16 + (lane & 15)) * LDB + kpart * 8];
      v4u lo = *reinterpret_cast<const v4u*>(base);                    // K = 8*kpart..+7 @ N=lane%16
      v4u hi = *reinterpret_cast<const v4u*>(base + 16);
      bF[ni] = __builtin_bit_cast(v16bf,
               __builtin_shufflevector(lo, hi, 0, 1, 2, 3, 4, 5, 6, 7));
    }
    #pragma unroll
    for (int mi = 0; mi < 2; ++mi)
      #pragma unroll
      for (int ni = 0; ni < 2; ++ni)
        acc[mi][ni] = __builtin_amdgcn_wmma_f32_16x16x32_bf16(
            false, aF[mi], false, bF[ni], (short)0, acc[mi][ni], false, false);
    __syncthreads();

    // ---- advance bank/dbase without division ----
    dbase += BK;
    if (dbase == A_COLS) {
      dbase = 0;
      ++bank;
      if (k0 + BK < kend) s = sel[a_r * NBANK + bank];
    }
  }

  // ---- epilogue: C lane layout -> f32 atomic accumulate ----
  const int cn = lane & 15;
  #pragma unroll
  for (int mi = 0; mi < 2; ++mi) {
    #pragma unroll
    for (int ni = 0; ni < 2; ++ni) {
      #pragma unroll
      for (int r = 0; r < 8; ++r) {
        const int grow = wm * 32 + mi * 16 + 8 * kpart + r;
        const int gcol = n0 + wn * 32 + ni * 16 + cn;
        float v = acc[mi][ni][r];
        if (kbeg == 0) {               // fold banked bias once
          float bs = 0.f;
          #pragma unroll
          for (int kb = 0; kb < NBANK; ++kb)
            bs += sel[grow * NBANK + kb] * bias[kb * N + gcol];
          v += bs;
        }
        unsafeAtomicAdd(&Cout[(size_t)grow * N + gcol], v);
      }
    }
  }
}

// Linear-attention read: per (b,m): q=elu(q)+1; num = matrix^T q; den = q.norm + 1e-9.
// 4 (b,m) pairs per 256-thread block; streams `matrix` coalesced + non-temporal.
__global__ __launch_bounds__(256) void attn_kernel(
    const float* __restrict__ q, const float* __restrict__ matrix,
    const float* __restrict__ normalizer, float* __restrict__ r)
{
  __shared__ float qs[4][64];
  __shared__ float red[4][64];
  const int g  = threadIdx.x >> 6;
  const int t  = threadIdx.x & 63;
  const int bm = blockIdx.x * 4 + g;           // b*128 + m

  const float qraw = q[(size_t)bm * 64 + t];
  const float qk = qraw > 0.f ? (qraw + 1.f) : __expf(qraw);   // elu(x)+1
  qs[g][t]  = qk;
  red[g][t] = qk * normalizer[(size_t)bm * 64 + t];
  __syncthreads();
  #pragma unroll
  for (int s = 32; s > 0; s >>= 1) {
    if (t < s) red[g][t] += red[g][t + s];
    __syncthreads();
  }
  const float denom = red[g][0] + 1e-9f;

  const float* mp = &matrix[(size_t)bm * 4096];  // (64 k) x (64 v)
  float num = 0.f;
  #pragma unroll 8
  for (int k = 0; k < 64; ++k)
    num = fmaf(__builtin_nontemporal_load(&mp[k * 64 + t]), qs[g][k], num);
  r[(size_t)bm * 64 + t] = num / denom;
}

__global__ void zero_kernel(float* __restrict__ p, int n) {
  int i = blockIdx.x * blockDim.x + threadIdx.x;
  if (i < n) p[i] = 0.f;
}

extern "C" void kernel_launch(void* const* d_in, const int* in_sizes, int n_in,
                              void* d_out, int out_size, void* d_ws, size_t ws_size,
                              hipStream_t stream) {
  const float* tensor     = (const float*)d_in[0];   // (128,1024)
  const float* matrix     = (const float*)d_in[1];   // (128,128,64,64)
  const float* normalizer = (const float*)d_in[2];   // (128,128,64)
  const float* sel        = (const float*)d_in[3];   // (128,8)
  const float* W_make     = (const float*)d_in[4];   // (8,1024,8192) -> (8192,8192)
  const float* b_make     = (const float*)d_in[5];   // (8,8192)
  const float* W_merge    = (const float*)d_in[6];   // (8,8192,1024) -> (65536,1024)
  const float* b_merge    = (const float*)d_in[7];   // (8,1024)
  float* out = (float*)d_out;                        // (128,1024)

  float* q = (float*)d_ws;                           // (128,8192) f32
  float* r = q + (size_t)128 * 8192;                 // (128,8192) f32

  zero_kernel<<<dim3((128 * 8192) / 256), 256, 0, stream>>>(q, 128 * 8192);
  zero_kernel<<<dim3((128 * 1024) / 256), 256, 0, stream>>>(out, 128 * 1024);

  // make-heads: M=128, N=8192, K=8*1024=8192, split-K x4 -> 512 WGs
  banked_gemm_kernel<1024><<<dim3(8192 / BN, 4), 256, 0, stream>>>(
      tensor, sel, W_make, b_make, q, 8192, 2048);

  // attention read: 16384 (b,m) pairs, 4 per block
  attn_kernel<<<dim3(16384 / 4), 256, 0, stream>>>(q, matrix, normalizer, r);

  // merge-heads: M=128, N=1024, K=8*8192=65536, split-K x32 -> 512 WGs
  banked_gemm_kernel<8192><<<dim3(1024 / BN, 32), 256, 0, stream>>>(
      r, sel, W_merge, b_merge, out, 1024, 2048);
}
